// MoEStage_21457656611374
// MI455X (gfx1250) — matrix operand order repaired and checked by reading, not compiled, not run on hardware
//
#include <hip/hip_runtime.h>
#include <hip/hip_bf16.h>

// Problem constants (from reference)
#define NTOK   8192
#define DMODEL 1024
#define NFEAT  32
#define NEXP   8
#define NPEF   8
#define DFE    128
#define DRH    256
#define DEH    512
#define DIN    (DMODEL + DFE)   // 1152

typedef __bf16 bf16_t;
typedef __attribute__((ext_vector_type(16))) __bf16 v16bf;
typedef __attribute__((ext_vector_type(8)))  __bf16 v8bf;
typedef __attribute__((ext_vector_type(8)))  float  v8f;

union BF16x16 { v16bf v; v8bf h[2]; };

// Scheduling fence: pin issue order  loads(k+1) | wmma(k) | loads(k+2) | wmma(k+1)
// so the waitcnt pass emits partial s_wait_loadcnt (verified: 0x12..0xa).
#define SCHED_FENCE() __builtin_amdgcn_sched_barrier(0)

// A-operand tile (16x32 bf16): per lane, K chunks {0..7,16..23} (lanes 0-15)
// or {8..15,24..31} (lanes 16-31). Caller applies the +0/+8 lane offset.
__device__ __forceinline__ v16bf load_a16(const bf16_t* p) {
  BF16x16 u;
  u.h[0] = *(const v8bf*)(p);
  u.h[1] = *(const v8bf*)(p + 16);
  return u.v;
}
// B-operand tile (32x16 bf16): per lane (column), K chunks {0..7,16..23}
// (lanes 0-15) or {8..15,24..31} (lanes 16-31). Caller applies +0/+8.
__device__ __forceinline__ v16bf load_b16(const bf16_t* p) {
  BF16x16 u;
  u.h[0] = *(const v8bf*)(p);
  u.h[1] = *(const v8bf*)(p + 16);
  return u.v;
}

__device__ __forceinline__ float gelu_tanh(float x) {
  float x3 = x * x * x;
  return 0.5f * x * (1.0f + tanhf(0.7978845608028654f * (x + 0.044715f * x3)));
}

#define WMMA_BF16(a, b, c) \
  __builtin_amdgcn_wmma_f32_16x16x32_bf16(false, (a), false, (b), (short)0, (c), false, false)

// ---------------------------------------------------------------------------
// Fused expert kernel (defined FIRST so the disasm snippet shows its loop).
// Block = 256 threads (8 waves), 32 tokens per block.
// Wave tile: 32 rows x 64 cols (2 A-tiles x 4 B-tiles = 8 WMMA per k-step).
// Branch-free pipelined K loops: hidden main loop (K 0..1023) + unrolled tail
// for hidden steps 30,31 and the 4 fe steps (K 1024..1151).
// Per expert: GEMM1 (1152->512) -> GELU -> bf16 h1 in LDS ->
// GEMM2 (512->1024) in 2 passes -> gate-scale -> out += (exclusive RMW).
// ---------------------------------------------------------------------------
#define H1_STRIDE 520   // 512 + 8 pad: LDS row stride = 260 dwords -> bank +4/row
#define NK2 (DEH / 32)  // 16 (even)

__global__ __launch_bounds__(256) void k_expert(
    const bf16_t* __restrict__ hiddenbf, const bf16_t* __restrict__ febf,
    const bf16_t* __restrict__ we1t, const bf16_t* __restrict__ we2t,
    const float* __restrict__ be1, const float* __restrict__ be2,
    const float* __restrict__ gatew, float* __restrict__ out) {
  __shared__ bf16_t h1[32 * H1_STRIDE];
  __shared__ float wgt[32];

  const int tid = threadIdx.x;
  const int lane = tid & 31;
  const int wave = tid >> 5;            // 0..7 : 64-col (GEMM1) / 128-col (GEMM2) slab
  const int rb0 = blockIdx.x * 32;
  const int lrow = lane & 15;
  const int hi = (lane >> 4) & 1;
  const int aoff = hi ? 8 : 0;
  const int boff = hi ? 8 : 0;

  const int r0 = rb0 + lrow;            // global row, row-tile 0
  const int r1 = rb0 + 16 + lrow;       // global row, row-tile 1

  for (int e = 0; e < NEXP; ++e) {
    if (tid < 32) wgt[tid] = gatew[(size_t)(rb0 + tid) * NEXP + e];

    // ---- GEMM1: h1[32,512] = gelu(A[32,1152] @ We1[e] + be1[e]) ----
    {
      const bf16_t* feb = febf + (size_t)e * NTOK * DFE;
      const bf16_t* w1b = we1t + ((size_t)e * DEH + 64 * wave + lrow) * DIN + boff;
      const bf16_t* ah0 = hiddenbf + (size_t)r0 * DMODEL + aoff;
      const bf16_t* ah1 = hiddenbf + (size_t)r1 * DMODEL + aoff;
      const bf16_t* af0 = feb + (size_t)r0 * DFE + aoff;
      const bf16_t* af1 = feb + (size_t)r1 * DFE + aoff;

      auto ldAh = [&](int kb, v16bf* a) {       // kb in [0,1024)
        a[0] = load_a16(ah0 + kb);
        a[1] = load_a16(ah1 + kb);
      };
      auto ldAf = [&](int kf, v16bf* a) {       // kf in [0,128)
        a[0] = load_a16(af0 + kf);
        a[1] = load_a16(af1 + kf);
      };
      auto ldB = [&](int kb, v16bf* b) {        // kb in [0,1152)
#pragma unroll
        for (int t = 0; t < 4; ++t) b[t] = load_b16(w1b + (size_t)(16 * t) * DIN + kb);
      };
      auto mm = [&](v8f (&c)[2][4], v16bf* a, v16bf* b) {
#pragma unroll
        for (int t = 0; t < 4; ++t) {
          c[0][t] = WMMA_BF16(a[0], b[t], c[0][t]);
          c[1][t] = WMMA_BF16(a[1], b[t], c[1][t]);
        }
      };

      v8f c1[2][4] = {};
      v16bf aC[2], bC[4], aN[2], bN[4];
      ldAh(0, aC); ldB(0, bC);
      SCHED_FENCE();
      for (int kt = 0; kt < 30; kt += 2) {      // consumes hidden steps 0..29
        ldAh((kt + 1) * 32, aN); ldB((kt + 1) * 32, bN);
        SCHED_FENCE();
        mm(c1, aC, bC);
        SCHED_FENCE();
        ldAh((kt + 2) * 32, aC); ldB((kt + 2) * 32, bC);
        SCHED_FENCE();
        mm(c1, aN, bN);
        SCHED_FENCE();
      }
      // tail: hidden steps 30,31 then fe steps 0..3 (K = 1024..1151)
      ldAh(31 * 32, aN); ldB(31 * 32, bN);
      SCHED_FENCE();
      mm(c1, aC, bC);                            // hidden step 30
      SCHED_FENCE();
      ldAf(0, aC); ldB(DMODEL + 0, bC);
      SCHED_FENCE();
      mm(c1, aN, bN);                            // hidden step 31
      SCHED_FENCE();
      ldAf(32, aN); ldB(DMODEL + 32, bN);
      SCHED_FENCE();
      mm(c1, aC, bC);                            // fe step 0
      SCHED_FENCE();
      ldAf(64, aC); ldB(DMODEL + 64, bC);
      SCHED_FENCE();
      mm(c1, aN, bN);                            // fe step 1
      SCHED_FENCE();
      ldAf(96, aN); ldB(DMODEL + 96, bN);
      SCHED_FENCE();
      mm(c1, aC, bC);                            // fe step 2
      SCHED_FENCE();
      mm(c1, aN, bN);                            // fe step 3

#pragma unroll
      for (int mt = 0; mt < 2; ++mt) {
#pragma unroll
        for (int t = 0; t < 4; ++t) {
#pragma unroll
          for (int r = 0; r < 8; ++r) {
            const int row = 16 * mt + r + 8 * hi;
            const int col = 64 * wave + 16 * t + lrow;
            float x = c1[mt][t][r] + be1[e * DEH + col];
            h1[row * H1_STRIDE + col] = (bf16_t)gelu_tanh(x);
          }
        }
      }
    }
    __syncthreads();   // h1 + wgt visible to all waves

    float wv[2][8];
#pragma unroll
    for (int mt = 0; mt < 2; ++mt)
#pragma unroll
      for (int r = 0; r < 8; ++r) wv[mt][r] = wgt[16 * mt + r + 8 * hi];

    // ---- GEMM2: out[32,1024] += w * (h1 @ We2[e] + be2[e]) ----
#pragma unroll
    for (int p = 0; p < 2; ++p) {
      const bf16_t* w2b =
          we2t + ((size_t)e * DMODEL + 128 * wave + 64 * p + lrow) * DEH + boff;

      auto ldA2 = [&](int kb, v16bf* a) {
        a[0] = load_a16(&h1[(size_t)lrow * H1_STRIDE + kb + aoff]);
        a[1] = load_a16(&h1[(size_t)(16 + lrow) * H1_STRIDE + kb + aoff]);
      };
      auto ldB2 = [&](int kb, v16bf* b) {
#pragma unroll
        for (int t = 0; t < 4; ++t) b[t] = load_b16(w2b + (size_t)(16 * t) * DEH + kb);
      };
      auto mm = [&](v8f (&c)[2][4], v16bf* a, v16bf* b) {
#pragma unroll
        for (int t = 0; t < 4; ++t) {
          c[0][t] = WMMA_BF16(a[0], b[t], c[0][t]);
          c[1][t] = WMMA_BF16(a[1], b[t], c[1][t]);
        }
      };

      v8f c2[2][4] = {};
      v16bf aC[2], bC[4], aN[2], bN[4];
      ldA2(0, aC); ldB2(0, bC);
      SCHED_FENCE();
      for (int kt = 0; kt < NK2 - 2; kt += 2) {  // consumes steps 0..13
        ldA2((kt + 1) * 32, aN); ldB2((kt + 1) * 32, bN);
        SCHED_FENCE();
        mm(c2, aC, bC);
        SCHED_FENCE();
        ldA2((kt + 2) * 32, aC); ldB2((kt + 2) * 32, bC);
        SCHED_FENCE();
        mm(c2, aN, bN);
        SCHED_FENCE();
      }
      ldA2((NK2 - 1) * 32, aN); ldB2((NK2 - 1) * 32, bN);
      SCHED_FENCE();
      mm(c2, aC, bC);                             // step 14
      SCHED_FENCE();
      mm(c2, aN, bN);                             // step 15

#pragma unroll
      for (int mt = 0; mt < 2; ++mt) {
#pragma unroll
        for (int t = 0; t < 4; ++t) {
#pragma unroll
          for (int r = 0; r < 8; ++r) {
            const int row = rb0 + 16 * mt + r + 8 * hi;
            const int col = 128 * wave + 64 * p + 16 * t + lrow;
            out[(size_t)row * DMODEL + col] +=
                wv[mt][r] * (c2[mt][t][r] + be2[e * DMODEL + col]);
          }
        }
      }
    }
    __syncthreads();   // protect h1/wgt before next expert overwrites
  }
}

// ---------------------------------------------------------------------------
// Router GEMM: rh[N,256] = gelu([hidden|fs] @ Wr1 + br1)   (WMMA bf16)
// Block: 256 threads = 8 waves as 2(M) x 4(N); block tile 32 x 256; K=1152.
// Same branch-free pipelined structure as k_expert GEMM1 (1 A-tile).
// ---------------------------------------------------------------------------
__global__ __launch_bounds__(256) void k_router(
    const bf16_t* __restrict__ hiddenbf, const bf16_t* __restrict__ fsbf,
    const bf16_t* __restrict__ wr1t, const float* __restrict__ br1,
    float* __restrict__ rh) {
  const int tid = threadIdx.x;
  const int lane = tid & 31;
  const int wave = tid >> 5;
  const int wm = wave >> 2, wn = wave & 3;
  const int rb0 = blockIdx.x * 32;
  const int lrow = lane & 15;
  const int hi = (lane >> 4) & 1;
  const int aoff = hi ? 8 : 0;
  const int boff = hi ? 8 : 0;

  const int arow = rb0 + 16 * wm + lrow;
  const bf16_t* ahp = hiddenbf + (size_t)arow * DMODEL + aoff;
  const bf16_t* afp = fsbf + (size_t)arow * DFE + aoff;
  const bf16_t* wrb = wr1t + ((size_t)(64 * wn) + lrow) * DIN + boff;

  auto ldB = [&](int kb, v16bf* b) {
#pragma unroll
    for (int t = 0; t < 4; ++t) b[t] = load_b16(wrb + (size_t)(16 * t) * DIN + kb);
  };
  auto mm = [&](v8f (&c)[4], v16bf a, v16bf* b) {
#pragma unroll
    for (int t = 0; t < 4; ++t) c[t] = WMMA_BF16(a, b[t], c[t]);
  };

  v8f c[4] = {};
  v16bf aC, bC[4], aN, bN[4];
  aC = load_a16(ahp); ldB(0, bC);
  SCHED_FENCE();
  for (int kt = 0; kt < 30; kt += 2) {
    aN = load_a16(ahp + (kt + 1) * 32); ldB((kt + 1) * 32, bN);
    SCHED_FENCE();
    mm(c, aC, bC);
    SCHED_FENCE();
    aC = load_a16(ahp + (kt + 2) * 32); ldB((kt + 2) * 32, bC);
    SCHED_FENCE();
    mm(c, aN, bN);
    SCHED_FENCE();
  }
  aN = load_a16(ahp + 31 * 32); ldB(31 * 32, bN);
  SCHED_FENCE();
  mm(c, aC, bC);                               // hidden step 30
  SCHED_FENCE();
  aC = load_a16(afp + 0); ldB(DMODEL + 0, bC);
  SCHED_FENCE();
  mm(c, aN, bN);                               // hidden step 31
  SCHED_FENCE();
  aN = load_a16(afp + 32); ldB(DMODEL + 32, bN);
  SCHED_FENCE();
  mm(c, aC, bC);                               // fs step 0
  SCHED_FENCE();
  aC = load_a16(afp + 64); ldB(DMODEL + 64, bC);
  SCHED_FENCE();
  mm(c, aN, bN);                               // fs step 1
  SCHED_FENCE();
  aN = load_a16(afp + 96); ldB(DMODEL + 96, bN);
  SCHED_FENCE();
  mm(c, aC, bC);                               // fs step 2
  SCHED_FENCE();
  mm(c, aN, bN);                               // fs step 3

#pragma unroll
  for (int t = 0; t < 4; ++t) {
#pragma unroll
    for (int r = 0; r < 8; ++r) {
      const int row = rb0 + 16 * wm + r + 8 * hi;
      const int col = 64 * wn + 16 * t + lrow;
      rh[(size_t)row * DRH + col] = gelu_tanh(c[t][r] + br1[col]);
    }
  }
}

// ---------------------------------------------------------------------------
// Prep kernels
// ---------------------------------------------------------------------------
__global__ void k_cvt_hidden(const float* __restrict__ h, bf16_t* __restrict__ hb,
                             float* __restrict__ out) {
  size_t i = (size_t)blockIdx.x * blockDim.x + threadIdx.x;
  if (i >= (size_t)NTOK * DMODEL) return;
  float v = h[i];
  hb[i] = (bf16_t)v;
  out[i] = v;              // residual init: out = hidden
}

// W: [B, K, N] row-major f32  ->  WT: [B, N, K] row-major bf16
__global__ void k_tpose(const float* __restrict__ W, bf16_t* __restrict__ WT,
                        int B, int K, int Nn) {
  size_t total = (size_t)B * K * Nn;
  size_t i = (size_t)blockIdx.x * blockDim.x + threadIdx.x;
  if (i >= total) return;
  size_t kn = (size_t)K * Nn;
  int b = (int)(i / kn);
  size_t r = i % kn;
  int k = (int)(r / Nn);
  int n = (int)(r % Nn);
  WT[((size_t)b * Nn + n) * K + k] = (bf16_t)W[i];
}

// fs[n,:] = feat[n, stage_idx] @ Wfs + bfs   (K=32)
// fe[e,n,:] = feat[n, expert_idx[e]] @ Wfe[e] + bfe[e]   (K=8)
__global__ void k_fs_fe(const float* __restrict__ feat, const float* __restrict__ Wfs,
                        const float* __restrict__ bfs, const float* __restrict__ Wfe,
                        const float* __restrict__ bfe, const int* __restrict__ expert_idx,
                        const int* __restrict__ stage_idx,
                        bf16_t* __restrict__ fsbf, bf16_t* __restrict__ febf) {
  int i = blockIdx.x * blockDim.x + threadIdx.x;      // over NTOK*DFE
  if (i >= NTOK * DFE) return;
  int n = i >> 7;
  int d = i & (DFE - 1);
  const float* frow = feat + (size_t)n * NFEAT;
  float acc = bfs[d];
#pragma unroll
  for (int k = 0; k < NFEAT; ++k) acc += frow[stage_idx[k]] * Wfs[k * DFE + d];
  fsbf[(size_t)n * DFE + d] = (bf16_t)acc;
#pragma unroll
  for (int e = 0; e < NEXP; ++e) {
    float a = bfe[e * DFE + d];
#pragma unroll
    for (int k = 0; k < NPEF; ++k)
      a += frow[expert_idx[e * NPEF + k]] * Wfe[(e * NPEF + k) * DFE + d];
    febf[((size_t)e * NTOK + n) * DFE + d] = (bf16_t)a;
  }
}

// ---------------------------------------------------------------------------
// Gating: logits = rh @ Wr2 + br2; top-2 softmax scattered dense (JAX ties).
// ---------------------------------------------------------------------------
__global__ void k_gate(const float* __restrict__ rh, const float* __restrict__ Wr2,
                       const float* __restrict__ br2, float* __restrict__ gatew) {
  int n = blockIdx.x * blockDim.x + threadIdx.x;
  if (n >= NTOK) return;
  float lg[NEXP];
#pragma unroll
  for (int e = 0; e < NEXP; ++e) lg[e] = br2[e];
  const float* r = rh + (size_t)n * DRH;
  for (int k = 0; k < DRH; ++k) {
    float rv = r[k];
#pragma unroll
    for (int e = 0; e < NEXP; ++e) lg[e] += rv * Wr2[k * NEXP + e];
  }
  int i1 = 0;
#pragma unroll
  for (int e = 1; e < NEXP; ++e) if (lg[e] > lg[i1]) i1 = e;   // lowest idx on ties
  int i2 = (i1 == 0) ? 1 : 0;
#pragma unroll
  for (int e = 0; e < NEXP; ++e)
    if (e != i1 && lg[e] > lg[i2]) i2 = e;
  float e2 = expf(lg[i2] - lg[i1]);
  float w1 = 1.0f / (1.0f + e2);
  float w2 = e2 * w1;
#pragma unroll
  for (int e = 0; e < NEXP; ++e) gatew[(size_t)n * NEXP + e] = 0.0f;
  gatew[(size_t)n * NEXP + i1] = w1;
  gatew[(size_t)n * NEXP + i2] = w2;
}

// ---------------------------------------------------------------------------
extern "C" void kernel_launch(void* const* d_in, const int* in_sizes, int n_in,
                              void* d_out, int out_size, void* d_ws, size_t ws_size,
                              hipStream_t stream) {
  const float* hidden = (const float*)d_in[0];
  const float* feat   = (const float*)d_in[1];
  const float* Wfs    = (const float*)d_in[2];
  const float* bfs    = (const float*)d_in[3];
  const float* Wr1    = (const float*)d_in[4];
  const float* br1    = (const float*)d_in[5];
  const float* Wr2    = (const float*)d_in[6];
  const float* br2    = (const float*)d_in[7];
  const float* Wfe    = (const float*)d_in[8];
  const float* bfe    = (const float*)d_in[9];
  const float* We1    = (const float*)d_in[10];
  const float* be1    = (const float*)d_in[11];
  const float* We2    = (const float*)d_in[12];
  const float* be2    = (const float*)d_in[13];
  const int* expert_idx = (const int*)d_in[14];
  const int* stage_idx  = (const int*)d_in[15];
  float* out = (float*)d_out;

  // Workspace carve-out (256B aligned slabs), ~63 MB total.
  char* ws = (char*)d_ws;
  auto alloc = [&](size_t bytes) {
    char* p = ws;
    ws += (bytes + 255) & ~(size_t)255;
    return p;
  };
  bf16_t* hiddenbf = (bf16_t*)alloc((size_t)NTOK * DMODEL * sizeof(bf16_t));
  bf16_t* fsbf     = (bf16_t*)alloc((size_t)NTOK * DFE * sizeof(bf16_t));
  bf16_t* febf     = (bf16_t*)alloc((size_t)NEXP * NTOK * DFE * sizeof(bf16_t));
  bf16_t* wr1t     = (bf16_t*)alloc((size_t)DRH * DIN * sizeof(bf16_t));
  bf16_t* we1t     = (bf16_t*)alloc((size_t)NEXP * DEH * DIN * sizeof(bf16_t));
  bf16_t* we2t     = (bf16_t*)alloc((size_t)NEXP * DMODEL * DEH * sizeof(bf16_t));
  float*  rh       = (float*)alloc((size_t)NTOK * DRH * sizeof(float));
  float*  gatew    = (float*)alloc((size_t)NTOK * NEXP * sizeof(float));

  const int T = 256;
  k_cvt_hidden<<<(NTOK * DMODEL + T - 1) / T, T, 0, stream>>>(hidden, hiddenbf, out);
  k_tpose<<<(1 * DIN * DRH + T - 1) / T, T, 0, stream>>>(Wr1, wr1t, 1, DIN, DRH);
  k_tpose<<<(NEXP * DIN * DEH + T - 1) / T, T, 0, stream>>>(We1, we1t, NEXP, DIN, DEH);
  k_tpose<<<(NEXP * DEH * DMODEL + T - 1) / T, T, 0, stream>>>(We2, we2t, NEXP, DEH, DMODEL);
  k_fs_fe<<<(NTOK * DFE + T - 1) / T, T, 0, stream>>>(feat, Wfs, bfs, Wfe, bfe,
                                                      expert_idx, stage_idx, fsbf, febf);
  k_router<<<NTOK / 32, T, 0, stream>>>(hiddenbf, fsbf, wr1t, br1, rh);
  k_gate<<<(NTOK + T - 1) / T, T, 0, stream>>>(rh, Wr2, br2, gatew);
  k_expert<<<NTOK / 32, T, 0, stream>>>(hiddenbf, febf, we1t, we2t, be1, be2, gatew, out);
}
